// QuantizedLinear_41240275976514
// MI455X (gfx1250) — matrix operand order repaired and checked by reading, not compiled, not run on hardware
//
#include <hip/hip_runtime.h>

// Problem constants (fixed by the reference setup_inputs()).
#define T_DIM 2048
#define I_DIM 4096
#define O_DIM 11008
#define GROUP 128

// Tiling.
#define BM 128
#define BN 128
#define BK 128          // == GROUP: one scale per (row, k-block)
#define LDSTRIDE 136    // padded LDS row stride in bf16 elements (128 + 8)

typedef __attribute__((ext_vector_type(16))) __bf16 v16bf;
typedef __attribute__((ext_vector_type(2)))  __bf16 v2bf;
typedef __attribute__((ext_vector_type(8)))  float  v8f;

// float2 -> packed bf16x2. Prefer the hardware v_cvt_pk_bf16_f32.
__device__ __forceinline__ unsigned int pack_bf16x2(float lo, float hi) {
#if __has_builtin(__builtin_amdgcn_cvt_pk_bf16_f32)
    union { v2bf v; unsigned int u; } t;
    t.v = __builtin_amdgcn_cvt_pk_bf16_f32(lo, hi);
    return t.u;
#else
    unsigned int ulo = __float_as_uint(lo);
    unsigned int uhi = __float_as_uint(hi);
    ulo = (ulo + 0x7FFFu + ((ulo >> 16) & 1u)) >> 16;
    uhi = (uhi + 0x7FFFu + ((uhi >> 16) & 1u)) >> 16;
    return ulo | (uhi << 16);
#endif
}

// Dequant one packed byte (2 nibbles) -> packed bf16x2: w = nib*s - 8s.
__device__ __forceinline__ unsigned int dq2(int q, float s, float nb) {
    const float lo = __builtin_fmaf((float)(q & 15), s, nb);
    const float hi = __builtin_fmaf((float)((q >> 4) & 15), s, nb);
    return pack_bf16x2(lo, hi);
}

// Load a 32-byte WMMA fragment (16 bf16) from two 16B LDS chunks.
__device__ __forceinline__ v16bf load_frag(const unsigned short* p0,
                                           const unsigned short* p1) {
    union { uint4 u[2]; v16bf v; } t;
    t.u[0] = *(const uint4*)p0;
    t.u[1] = *(const uint4*)p1;
    return t.v;
}

// ============================================================================
// Pre-pass 1: x fp32 -> bf16 (bandwidth-bound, done once instead of 86x).
// ============================================================================
__global__ __launch_bounds__(256) void x_to_bf16(const float* __restrict__ x,
                                                 unsigned short* __restrict__ xbf) {
    const int t = blockIdx.x * 256 + threadIdx.x;   // [0, T*I/8)
    const float4 a = *(const float4*)(x + (size_t)t * 8);
    const float4 b = *(const float4*)(x + (size_t)t * 8 + 4);
    uint4 p;
    p.x = pack_bf16x2(a.x, a.y);
    p.y = pack_bf16x2(a.z, a.w);
    p.z = pack_bf16x2(b.x, b.y);
    p.w = pack_bf16x2(b.z, b.w);
    *(uint4*)(xbf + (size_t)t * 8) = p;
}

// ============================================================================
// Pre-pass 2: W int4 -> bf16 dequant (done once instead of 16x).
// ============================================================================
__global__ __launch_bounds__(256) void w_dequant_bf16(
    const int* __restrict__ qw, const float* __restrict__ scales,
    unsigned short* __restrict__ wbf) {
    const int t   = blockIdx.x * 256 + threadIdx.x; // [0, O * 512)
    const int row = t >> 9;                         // output feature
    const int c   = t & 511;                        // int4-chunk within row
    const int4 q  = *(const int4*)(qw + (size_t)row * (I_DIM / 2) + c * 4);
    const float s  = scales[(size_t)row * (I_DIM / GROUP) + (c >> 4)];
    const float nb = -8.0f * s;
    uint4 p;
    p.x = dq2(q.x, s, nb);
    p.y = dq2(q.y, s, nb);
    p.z = dq2(q.z, s, nb);
    p.w = dq2(q.w, s, nb);
    *(uint4*)(wbf + (size_t)row * I_DIM + c * 8) = p;
}

// ============================================================================
// Main bf16 GEMM: out = xbf @ wbf^T + bias.
// 128 threads = 4 waves in a 2x2 grid; each wave owns a 64x64 output tile
// (16 WMMA accum tiles) -> 1.0 ds_load_b128 per v_wmma. Tiles are DMA'd into
// double-buffered LDS with global_load_async_to_lds_b128 (ASYNCcnt).
// ============================================================================
__global__ __launch_bounds__(128) void gemm_bf16_async(
    const unsigned short* __restrict__ xbf,   // [T, I] bf16
    const unsigned short* __restrict__ wbf,   // [O, I] bf16
    const float* __restrict__ bias,           // [O] fp32
    float* __restrict__ out) {                // [T, O] fp32

    __shared__ __align__(16) unsigned short shA[2][BM * LDSTRIDE];
    __shared__ __align__(16) unsigned short shB[2][BN * LDSTRIDE];

    const int tid  = threadIdx.x;
    const int lane = tid & 31;
    const int m16  = lane & 15;   // row (A) / col (B/D) within 16
    const int hh   = lane >> 4;   // lane half
    const int wave  = tid >> 5;          // 4 waves
    const int waveM = (wave & 1) * 64;   // 2 waves along M
    const int waveN = (wave >> 1) * 64;  // 2 waves along N

    const int blockN = blockIdx.x * BN;
    const int blockM = blockIdx.y * BM;

    // DMA one k-block into buf: each tile is 128 rows x 256 B = 2048 x 16B
    // chunks; 128 threads -> 16 chunks/thread per tile (A and B).
    auto issue_async = [&](int kt, int buf) {
        const unsigned long long abase =
            (unsigned long long)(const void*)(xbf + (size_t)blockM * I_DIM + kt * BK);
        const unsigned long long bbase =
            (unsigned long long)(const void*)(wbf + (size_t)blockN * I_DIM + kt * BK);
        #pragma unroll
        for (int i = 0; i < 16; ++i) {
            const int idx = tid + i * 128;
            const int row = idx >> 4;
            const int c16 = idx & 15;
            const unsigned goff = (unsigned)(row * (I_DIM * 2) + c16 * 16); // bytes
            const unsigned lA =
                (unsigned)(unsigned long long)&shA[buf][row * LDSTRIDE + c16 * 8];
            const unsigned lB =
                (unsigned)(unsigned long long)&shB[buf][row * LDSTRIDE + c16 * 8];
            asm volatile("global_load_async_to_lds_b128 %0, %1, %2"
                         :: "v"(lA), "v"(goff), "s"(abase) : "memory");
            asm volatile("global_load_async_to_lds_b128 %0, %1, %2"
                         :: "v"(lB), "v"(goff), "s"(bbase) : "memory");
        }
    };

    v8f acc[4][4] = {};  // 64 x 64 per wave = 4x4 tiles of 16x16

    // WMMA over one k-block: 4 steps of K=32, 16 WMMAs per step.
    auto compute = [&](int buf) {
        #pragma unroll
        for (int kk = 0; kk < BK; kk += 32) {
            v16bf afrag[4], bfrag[4];
            // A 16x32 bf16: elems 0..7 -> K = kk + h*8 + e; 8..15 -> +16.
            #pragma unroll
            for (int i = 0; i < 4; ++i) {
                const unsigned short* base =
                    &shA[buf][(waveM + i * 16 + m16) * LDSTRIDE + kk + hh * 8];
                afrag[i] = load_frag(base, base + 16);
            }
            // B 32x16 bf16: elems 0..7 -> K = kk + h*8 + e; 8..15 -> +16.
            #pragma unroll
            for (int j = 0; j < 4; ++j) {
                const unsigned short* base =
                    &shB[buf][(waveN + j * 16 + m16) * LDSTRIDE + kk + hh * 8];
                bfrag[j] = load_frag(base, base + 16);
            }
            #pragma unroll
            for (int i = 0; i < 4; ++i) {
                #pragma unroll
                for (int j = 0; j < 4; ++j) {
                    acc[i][j] = __builtin_amdgcn_wmma_f32_16x16x32_bf16(
                        false, afrag[i], false, bfrag[j],
                        (short)0, acc[i][j], false, false);
                }
            }
        }
    };

    const int KB = I_DIM / BK;  // 32 k-blocks

    // Prologue: DMA buffer 0, wait, publish.
    issue_async(0, 0);
    asm volatile("s_wait_asynccnt 0x0" ::: "memory");
    __syncthreads();

    // Pipelined loop: DMA for kt+1 runs under the WMMA stream for kt.
    for (int kt = 0; kt < KB; ++kt) {
        const int cur = kt & 1;
        if (kt + 1 < KB) issue_async(kt + 1, cur ^ 1);
        compute(cur);
        asm volatile("s_wait_asynccnt 0x0" ::: "memory");
        __syncthreads();
    }

    // Epilogue: C/D layout lane={half h, col m16}, VGPR r = row h*8+r.
    #pragma unroll
    for (int j = 0; j < 4; ++j) {
        const int colg = blockN + waveN + j * 16 + m16;
        const float bj = bias[colg];
        #pragma unroll
        for (int i = 0; i < 4; ++i) {
            const int rowg = blockM + waveM + i * 16 + hh * 8;
            #pragma unroll
            for (int r = 0; r < 8; ++r) {
                out[(size_t)(rowg + r) * O_DIM + colg] = acc[i][j][r] + bj;
            }
        }
    }
}

// ============================================================================
// Fallback (no workspace): fused dequant GEMM, register-staged pipeline.
// ============================================================================
__global__ __launch_bounds__(256) void q4_gemm_fallback(
    const float* __restrict__ x, const int* __restrict__ qw,
    const float* __restrict__ scales, const float* __restrict__ bias,
    float* __restrict__ out) {

    __shared__ __align__(16) unsigned short shA[2][BM * LDSTRIDE];
    __shared__ __align__(16) unsigned short shB[2][BN * LDSTRIDE];

    const int tid  = threadIdx.x;
    const int lane = tid & 31;
    const int m16  = lane & 15;
    const int hh   = lane >> 4;
    const int wave  = tid >> 5;
    const int waveM = (wave & 1) * 64;
    const int waveN = (wave >> 1) * 32;
    const int blockN = blockIdx.x * BN;
    const int blockM = blockIdx.y * BM;

    float4 xr[16];
    int4   qr[8];
    float  sr[8];

    auto load_globals = [&](int kt) {
        const int k0 = kt * BK;
        const float* xg = x + (size_t)blockM * I_DIM + k0;
        #pragma unroll
        for (int i = 0; i < 16; ++i) {
            const int idx = tid + i * 256;
            xr[i] = *(const float4*)(xg + (size_t)(idx >> 5) * I_DIM + (idx & 31) * 4);
        }
        const int*   qg = qw + (size_t)blockN * (I_DIM / 2) + (k0 >> 1);
        const float* sg = scales + (size_t)blockN * (I_DIM / GROUP) + kt;
        #pragma unroll
        for (int i = 0; i < 8; ++i) {
            const int idx = tid + i * 256;
            const int row = idx >> 4;
            qr[i] = *(const int4*)(qg + (size_t)row * (I_DIM / 2) + (idx & 15) * 4);
            sr[i] = sg[(size_t)row * (I_DIM / GROUP)];
        }
    };

    auto store_lds = [&](int buf) {
        #pragma unroll
        for (int i = 0; i < 16; ++i) {
            const int idx = tid + i * 256;
            uint2 p;
            p.x = pack_bf16x2(xr[i].x, xr[i].y);
            p.y = pack_bf16x2(xr[i].z, xr[i].w);
            *(uint2*)(&shA[buf][(idx >> 5) * LDSTRIDE + (idx & 31) * 4]) = p;
        }
        #pragma unroll
        for (int i = 0; i < 8; ++i) {
            const int idx = tid + i * 256;
            const float s  = sr[i];
            const float nb = -8.0f * s;
            uint4 p;
            p.x = dq2(qr[i].x, s, nb);
            p.y = dq2(qr[i].y, s, nb);
            p.z = dq2(qr[i].z, s, nb);
            p.w = dq2(qr[i].w, s, nb);
            *(uint4*)(&shB[buf][(idx >> 4) * LDSTRIDE + (idx & 15) * 8]) = p;
        }
    };

    v8f acc[4][2] = {};

    auto compute = [&](int buf) {
        #pragma unroll
        for (int kk = 0; kk < BK; kk += 32) {
            v16bf afrag[4], bfrag[2];
            #pragma unroll
            for (int i = 0; i < 4; ++i) {
                const unsigned short* base =
                    &shA[buf][(waveM + i * 16 + m16) * LDSTRIDE + kk + hh * 8];
                afrag[i] = load_frag(base, base + 16);
            }
            #pragma unroll
            for (int j = 0; j < 2; ++j) {
                const unsigned short* base =
                    &shB[buf][(waveN + j * 16 + m16) * LDSTRIDE + kk + hh * 8];
                bfrag[j] = load_frag(base, base + 16);
            }
            #pragma unroll
            for (int i = 0; i < 4; ++i)
                #pragma unroll
                for (int j = 0; j < 2; ++j)
                    acc[i][j] = __builtin_amdgcn_wmma_f32_16x16x32_bf16(
                        false, afrag[i], false, bfrag[j],
                        (short)0, acc[i][j], false, false);
        }
    };

    const int KB = I_DIM / BK;
    load_globals(0);
    store_lds(0);
    __syncthreads();
    for (int kt = 0; kt < KB; ++kt) {
        const int cur = kt & 1;
        if (kt + 1 < KB) load_globals(kt + 1);
        compute(cur);
        if (kt + 1 < KB) store_lds(cur ^ 1);
        __syncthreads();
    }

    #pragma unroll
    for (int j = 0; j < 2; ++j) {
        const int colg = blockN + waveN + j * 16 + m16;
        const float bj = bias[colg];
        #pragma unroll
        for (int i = 0; i < 4; ++i) {
            const int rowg = blockM + waveM + i * 16 + hh * 8;
            #pragma unroll
            for (int r = 0; r < 8; ++r) {
                out[(size_t)(rowg + r) * O_DIM + colg] = acc[i][j][r] + bj;
            }
        }
    }
}

extern "C" void kernel_launch(void* const* d_in, const int* in_sizes, int n_in,
                              void* d_out, int out_size, void* d_ws, size_t ws_size,
                              hipStream_t stream) {
    (void)in_sizes; (void)n_in; (void)out_size;
    const float* x      = (const float*)d_in[0];
    const int*   qw     = (const int*)d_in[1];
    const float* scales = (const float*)d_in[2];
    const float* bias   = (const float*)d_in[3];
    float* out = (float*)d_out;

    const size_t XBF_BYTES = (size_t)T_DIM * I_DIM * 2;   // 16 MB
    const size_t WBF_BYTES = (size_t)O_DIM * I_DIM * 2;   // 90 MB
    dim3 grid(O_DIM / BN, T_DIM / BM);                    // 86 x 16

    if (ws_size >= XBF_BYTES + WBF_BYTES) {
        unsigned short* xbf = (unsigned short*)d_ws;
        unsigned short* wbf = (unsigned short*)((char*)d_ws + XBF_BYTES);
        x_to_bf16<<<(T_DIM * I_DIM / 8) / 256, 256, 0, stream>>>(x, xbf);
        w_dequant_bf16<<<(O_DIM * (I_DIM / 8)) / 256, 256, 0, stream>>>(qw, scales, wbf);
        gemm_bf16_async<<<grid, 128, 0, stream>>>(xbf, wbf, bias, out);
    } else {
        q4_gemm_fallback<<<grid, 256, 0, stream>>>(x, qw, scales, bias, out);
    }
}
